// REDE_INFER_52810917872258
// MI455X (gfx1250) — matrix operand order repaired and checked by reading, not compiled, not run on hardware
//
#include <hip/hip_runtime.h>
#include <math.h>

typedef __attribute__((ext_vector_type(16))) _Float16 v16h;
typedef __attribute__((ext_vector_type(8)))  _Float16 v8h;
typedef __attribute__((ext_vector_type(2)))  __fp16   v2fp;   // cvt_pkrtz result type
typedef __attribute__((ext_vector_type(8)))  float    v8f;
typedef __attribute__((ext_vector_type(4)))  float    v4f;

#define LDS_PITCH 72   // halves per row (144 B: 16-B aligned, staggers banks)

struct Decays {
    float inv[7];        // inv[0]=1 unused; inv[1..6] = 1/decay_i
    float final_scale;   // 1/sqrt(7)
};

union V16U { v16h v; v8h h[2]; unsigned int u[8]; };

// pack 8 f32 -> 4 u32 holding f16 pairs (element 2j -> [15:0], 2j+1 -> [31:16])
static __device__ __forceinline__ void pack8_f16(const v8f& p, unsigned int o[4]) {
    #pragma unroll
    for (int j = 0; j < 4; ++j) {
        union { v2fp h; unsigned int u; } t;
        t.h = __builtin_amdgcn_cvt_pkrtz(p[2 * j], p[2 * j + 1]);   // v_cvt_pk_rtz_f16_f32
        o[j] = t.u;
    }
}

// swap a dword with the other 16-lane half (identity mapping): v_permlanex16_b32
static __device__ __forceinline__ unsigned int halfwave_swap(unsigned int x) {
    return (unsigned int)__builtin_amdgcn_permlanex16(
        0, (int)x, 0x76543210, (int)0xfedcba98, false, false);
}

__global__ __launch_bounds__(128)
void REDE_powser_kernel(const float* __restrict__ M, float* __restrict__ out, Decays dk) {
    __shared__ _Float16 Mr[64 * LDS_PITCH];  // row-major f16(M)
    __shared__ _Float16 Mt[64 * LDS_PITCH];  // transposed f16(M): Mt[col][row]

    const int tid  = threadIdx.x;      // 0..127
    const int lane = tid & 31;
    const int w    = tid >> 5;         // wave = 16-column stripe (nt) of Q
    const int g    = lane >> 4;        // half-wave group
    const int ln   = lane & 15;

    const float* Mb = M   + (size_t)blockIdx.x * 4096;
    float*       Ob = out + (size_t)blockIdx.x * 4096;

    // ---- stage f16(M) into LDS, row-major and transposed ----
    #pragma unroll
    for (int k = 0; k < 8; ++k) {
        int idx4 = tid + k * 128;            // float4 index 0..1023
        int e    = idx4 * 4;
        int row  = e >> 6, col = e & 63;
        v4f v = ((const v4f*)Mb)[idx4];
        union { v2fp h2[2]; _Float16 h[4]; unsigned long long u; } pk;
        pk.h2[0] = __builtin_amdgcn_cvt_pkrtz(v[0], v[1]);
        pk.h2[1] = __builtin_amdgcn_cvt_pkrtz(v[2], v[3]);
        *(unsigned long long*)&Mr[row * LDS_PITCH + col] = pk.u;     // 8-B store
        #pragma unroll
        for (int j = 0; j < 4; ++j)
            Mt[(col + j) * LDS_PITCH + row] = pk.h[j];               // transposed
    }
    __syncthreads();

    // ---- A-fragments of M^T (constant for the whole chain), from Mt ----
    // A layout: lane -> row m'=mt*16+L%16; elem e -> K=kt*32+g*16+e
    v16h Af[8];                          // [mt*2 + kt]
    #pragma unroll
    for (int mt = 0; mt < 4; ++mt) {
        #pragma unroll
        for (int kt = 0; kt < 2; ++kt) {
            const int base = (mt * 16 + ln) * LDS_PITCH + kt * 32;
            V16U a;
            a.h[0] = *(const v8h*)&Mt[base +     g * 16];
            a.h[1] = *(const v8h*)&Mt[base + 8 + g * 16];
            Af[mt * 2 + kt] = a.v;
        }
    }

    // ---- B-fragments of Q1 = M^T for this wave's column stripe, from Mr ----
    // B layout: lane -> column n=w*16+L%16; elem e -> K=kt*32+g*16+e; B[k][n]=M[n][k]
    v16h Bf[2];
    #pragma unroll
    for (int kt = 0; kt < 2; ++kt) {
        const int base = (w * 16 + ln) * LDS_PITCH + kt * 32 + g * 16;
        V16U b;
        b.h[0] = *(const v8h*)&Mr[base];
        b.h[1] = *(const v8h*)&Mr[base + 8];
        Bf[kt] = b.v;
    }

    // ---- acc = Q1 = M^T (exact f32): lane holds Q[mt*16+g*8+r][w*16+ln] = M[w*16+ln][...]
    v8f acc[4];
    #pragma unroll
    for (int mt = 0; mt < 4; ++mt)
        acc[mt] = *(const v8f*)&Mb[(w * 16 + ln) * 64 + mt * 16 + g * 8];

    // ---- chained power series, fully register-resident: Q_{i+1} = (M^T @ Q_i) * inv_decay
    for (int step = 1; step < 7; ++step) {
        v8f raw[4];
        #pragma unroll
        for (int mt = 0; mt < 4; ++mt) {
            v8f c = {};
            c = __builtin_amdgcn_wmma_f32_16x16x32_f16(false, Af[mt * 2 + 0], false, Bf[0],
                                                       (short)0, c, false, false);
            c = __builtin_amdgcn_wmma_f32_16x16x32_f16(false, Af[mt * 2 + 1], false, Bf[1],
                                                       (short)0, c, false, false);
            raw[mt] = c;
        }

        const float s = dk.inv[step];
        unsigned int pk[4][4];
        #pragma unroll
        for (int mt = 0; mt < 4; ++mt) {
            v8f p = raw[mt] * s;
            acc[mt] += p;
            pack8_f16(p, pk[mt]);
        }

        if (step != 6) {
            // Rebuild next-step B from our own D tiles. Fragment kt draws from tiles
            // 2kt (g=0 half of K) and 2kt+1 (g=1 half). Each lane already owns half the
            // rows; the other half lives in the partner lane (lane^16). Send exactly the
            // dword the partner needs (t), swap once, place by half-wave group:
            //   elems 0-7 : g=0 -> own lo,          g=1 -> partner hi (swapped)
            //   elems 8-15: g=0 -> partner lo (sw), g=1 -> own hi
            #pragma unroll
            for (int kt = 0; kt < 2; ++kt) {
                V16U b;
                #pragma unroll
                for (int j = 0; j < 4; ++j) {
                    unsigned int lo = pk[2 * kt][j];
                    unsigned int hi = pk[2 * kt + 1][j];
                    unsigned int t  = (g == 0) ? hi : lo;     // what partner needs
                    unsigned int x  = halfwave_swap(t);       // what partner sent us
                    b.u[j]     = (g == 0) ? lo : x;
                    b.u[4 + j] = (g == 0) ? x  : hi;
                }
                Bf[kt] = b.v;
            }
        }
    }

    // ---- write P = Q^T scaled by 1/sqrt(7): 32 contiguous bytes per lane per tile ----
    #pragma unroll
    for (int mt = 0; mt < 4; ++mt) {
        v8f o = acc[mt] * dk.final_scale;
        *(v8f*)&Ob[(w * 16 + ln) * 64 + mt * 16 + g * 8] = o;
    }
}

static void compute_decays(Decays& dk) {
    static const double parms[7][7] = {
        {1}, {0, 1}, {8, 1, 1}, {16, 25, 3, 1}, {260, 153, 65, 6, 1},
        {1856, 21, 629, 137, 10, 1}, {26240, 26603, 11253, 1990, 263, 15, 1}};
    static const int lens[7] = {1, 2, 3, 4, 5, 6, 7};
    double var[7];
    for (int n = 1; n <= 7; ++n) {
        double res = 0.0;
        for (int i = 0; i < lens[n - 1]; ++i) {
            double x = parms[n - 1][i];
            for (int j = 0; j < i; ++j) x *= (64.0 - (double)j);
            res += x;
        }
        var[n - 1] = res;
    }
    dk.inv[0] = 1.0f;
    for (int i = 1; i < 7; ++i) dk.inv[i] = (float)(1.0 / sqrt(var[i] / var[i - 1]));
    dk.final_scale = (float)(1.0 / sqrt(7.0));
}

extern "C" void kernel_launch(void* const* d_in, const int* in_sizes, int n_in,
                              void* d_out, int out_size, void* d_ws, size_t ws_size,
                              hipStream_t stream) {
    (void)n_in; (void)d_ws; (void)ws_size; (void)out_size;
    const float* M  = (const float*)d_in[0];
    float*      out = (float*)d_out;
    const int batches = in_sizes[0] / 4096;   // 8192 rows of 64*64

    Decays dk;
    compute_decays(dk);

    REDE_powser_kernel<<<batches, 128, 0, stream>>>(M, out, dk);
}
